// ScaleDotProductAttention_39900246180154
// MI455X (gfx1250) — matrix operand order repaired and checked by reading, not compiled, not run on hardware
//
#include <hip/hip_runtime.h>

typedef __attribute__((ext_vector_type(16))) _Float16 v16h;
typedef __attribute__((ext_vector_type(8)))  float    v8f;
typedef __attribute__((ext_vector_type(4)))  float    f4;

#define B_DIM  4
#define H_DIM  16
#define S_DIM  2048
#define DK_DIM 64
#define TI     32   // i-rows per workgroup (2 x 16-row WMMA subtiles)
#define TJ     32   // j-cols per inner iteration
#define LPITCH 36   // LDS row pitch in floats: 32 data + 4 pad (16B-aligned, conflict-free)

// exp(x * 1/sqrt(64)) == exp2(x * 0.125 * log2(e)) -- fold scale into one constant
#define SCALE_LOG2E 0.18033688011112042f

// One workgroup: (batch b, 32-row i-tile) x all 16 heads. 16 waves, wave h owns head h.
// Streams j in tiles of 32. Cross-head softmax (axis=1 of [B,H,S,S]) via LDS exchange.
// out = attn @ V accumulated in registers across the whole j loop.
__global__ __launch_bounds__(512, 1)
void sdpa_headsoftmax_fused(const float* __restrict__ qg,
                            const float* __restrict__ kg,
                            const float* __restrict__ vg,
                            float* __restrict__ outg,
                            float* __restrict__ attng)
{
    __shared__ float expbuf[H_DIM][TI][LPITCH];   // 72 KB: exp2(score*c), per head 32x32 tile
    __shared__ float sumbuf[TI][LPITCH];          // 4.5 KB: sum over heads per (i,j)

    const int tid   = threadIdx.x;
    const int lane  = tid & 31;
    const int h     = tid >> 5;       // head == wave id
    const int halfL = lane & 15;      // 0..15
    const int hiL   = lane >> 4;      // 0 or 1

    const int b  = blockIdx.x >> 6;            // S/32 = 64 i-tiles per batch
    const int i0 = (blockIdx.x & 63) << 5;

    const size_t bh = (size_t)b * H_DIM + h;
    const float* kp = kg + bh * (size_t)(S_DIM * DK_DIM);  // reshape view: [64][2048] row-major
    const float* vp = vg + bh * (size_t)(S_DIM * DK_DIM);  // [2048][64]

    // ---- Q A-operands: 2 i-subtiles x (16x64 -> two 16x32 f16 A tiles), loaded once ----
    // lane<16 (row=halfL): a[0..7]=K 0..7, a[8..15]=K 16..23 ; lane>=16: a[0..7]=K 8..15, a[8..15]=K 24..31
    v16h aq[2][2];
#pragma unroll
    for (int isub = 0; isub < 2; ++isub) {
        const float* qp = qg + (bh * S_DIM + (size_t)(i0 + isub * 16 + halfL)) * DK_DIM;
#pragma unroll
        for (int t = 0; t < 8; ++t) {
            aq[isub][0][t]     = (_Float16)qp[ 0 + hiL * 8 + t];
            aq[isub][0][8 + t] = (_Float16)qp[16 + hiL * 8 + t];
            aq[isub][1][t]     = (_Float16)qp[32 + hiL * 8 + t];
            aq[isub][1][8 + t] = (_Float16)qp[48 + hiL * 8 + t];
        }
    }

    v8f oacc[2][4];
#pragma unroll
    for (int isub = 0; isub < 2; ++isub)
#pragma unroll
        for (int n = 0; n < 4; ++n) oacc[isub][n] = (v8f){};

    const int cb0 = hiL * 8;        // this lane's A-layout column runs
    const int cb1 = 16 + hiL * 8;
    const int q0  = cb0 >> 2;       // float4 indices within an LDS row
    const int q1  = cb1 >> 2;

    // cooperative-reduction assignment: 512 threads cover 32 rows x 32 cols, 2 cols each
    const int br = tid >> 4;          // 0..31
    const int bc = (tid & 15) * 2;    // 0,2,...,30

#pragma unroll 1
    for (int j0 = 0; j0 < S_DIM; j0 += TJ) {
        // ---- phase A: score tiles + exp into LDS. K B-operands shared by both i-subtiles ----
#pragma unroll
        for (int jj = 0; jj < TJ; jj += 16) {
            const int jc = j0 + jj + halfL;
            // B layout (32x16 f16): lane = column; lanes 0-15 hold K 0..7,16..23, lanes 16-31 K 8..15,24..31
            v16h bk0, bk1;
#pragma unroll
            for (int kk = 0; kk < 16; ++kk) {
                bk0[kk] = (_Float16)kp[( 0 + hiL * 8 + (kk >> 3) * 16 + (kk & 7)) * S_DIM + jc];
                bk1[kk] = (_Float16)kp[(32 + hiL * 8 + (kk >> 3) * 16 + (kk & 7)) * S_DIM + jc];
            }
#pragma unroll
            for (int isub = 0; isub < 2; ++isub) {
                v8f c = {};
                c = __builtin_amdgcn_wmma_f32_16x16x32_f16(false, aq[isub][0], false, bk0,
                                                           (short)0, c, false, false);
                c = __builtin_amdgcn_wmma_f32_16x16x32_f16(false, aq[isub][1], false, bk1,
                                                           (short)0, c, false, false);
                // C layout: c[g] <-> (row = hiL*8+g, col = halfL).
                // Raw v_exp_f32 (arg is ~[-3,3]: no subnormal fixup needed) with folded scale.
#pragma unroll
                for (int g = 0; g < 8; ++g)
                    expbuf[h][isub * 16 + hiL * 8 + g][jj + halfL] =
                        __builtin_amdgcn_exp2f(c[g] * SCALE_LOG2E);
            }
        }
        __syncthreads();

        // ---- phase B: cooperative cross-head sums; each thread: 1 row, 2 cols ----
        {
            float s0 = 0.0f, s1 = 0.0f;
#pragma unroll
            for (int hh = 0; hh < H_DIM; ++hh) {
                float2 p = *(const float2*)&expbuf[hh][br][bc];
                s0 += p.x; s1 += p.y;
            }
            sumbuf[br][bc]     = s0;
            sumbuf[br][bc + 1] = s1;
        }
        __syncthreads();

        // ---- phase C: normalize at A-layout positions, write attn (NT), build A operand ----
        v16h aatt[2];
#pragma unroll
        for (int isub = 0; isub < 2; ++isub) {
            const int r = isub * 16 + halfL;
            const float4* e4 = reinterpret_cast<const float4*>(&expbuf[h][r][0]);
            const float4* s4 = reinterpret_cast<const float4*>(&sumbuf[r][0]);
            float ev[16], sm[16];
            *(float4*)&ev[0] = e4[q0]; *(float4*)&ev[4]  = e4[q0 + 1];
            *(float4*)&ev[8] = e4[q1]; *(float4*)&ev[12] = e4[q1 + 1];
            *(float4*)&sm[0] = s4[q0]; *(float4*)&sm[4]  = s4[q0 + 1];
            *(float4*)&sm[8] = s4[q1]; *(float4*)&sm[12] = s4[q1 + 1];

            // raw v_rcp_f32 + v_mul: sums are well inside normal range, 1-ulp rcp is fine
            float att[16];
#pragma unroll
            for (int t = 0; t < 16; ++t) att[t] = ev[t] * __builtin_amdgcn_rcpf(sm[t]);

            // non-temporal: don't let the 1.07 GB attn stream evict L2-resident K/V
            float* arow = attng + (bh * S_DIM + (size_t)(i0 + r)) * S_DIM + j0;
            __builtin_nontemporal_store((f4){att[0],  att[1],  att[2],  att[3]},  (f4*)(arow + cb0));
            __builtin_nontemporal_store((f4){att[4],  att[5],  att[6],  att[7]},  (f4*)(arow + cb0 + 4));
            __builtin_nontemporal_store((f4){att[8],  att[9],  att[10], att[11]}, (f4*)(arow + cb1));
            __builtin_nontemporal_store((f4){att[12], att[13], att[14], att[15]}, (f4*)(arow + cb1 + 4));

#pragma unroll
            for (int t = 0; t < 16; ++t) aatt[isub][t] = (_Float16)att[t];
        }

        // ---- out += attn(16x32) @ V(32x64); V B-operand shared by both i-subtiles ----
#pragma unroll
        for (int n = 0; n < 4; ++n) {
            v16h bv;
#pragma unroll
            for (int kk = 0; kk < 16; ++kk)
                bv[kk] = (_Float16)vp[(size_t)(j0 + hiL * 8 + (kk >> 3) * 16 + (kk & 7)) * DK_DIM + n * 16 + halfL];
            oacc[0][n] = __builtin_amdgcn_wmma_f32_16x16x32_f16(false, aatt[0], false, bv,
                                                                (short)0, oacc[0][n], false, false);
            oacc[1][n] = __builtin_amdgcn_wmma_f32_16x16x32_f16(false, aatt[1], false, bv,
                                                                (short)0, oacc[1][n], false, false);
        }
        __syncthreads();   // all LDS reads done before next tile overwrites
    }

    // ---- store out tiles (2 x 16x64 f32), non-temporal (never re-read) ----
#pragma unroll
    for (int isub = 0; isub < 2; ++isub)
#pragma unroll
        for (int n = 0; n < 4; ++n)
#pragma unroll
            for (int g = 0; g < 8; ++g)
                __builtin_nontemporal_store(
                    oacc[isub][n][g],
                    outg + (bh * S_DIM + (size_t)(i0 + isub * 16 + hiL * 8 + g)) * DK_DIM
                         + n * 16 + halfL);
}

extern "C" void kernel_launch(void* const* d_in, const int* in_sizes, int n_in,
                              void* d_out, int out_size, void* d_ws, size_t ws_size,
                              hipStream_t stream) {
    (void)in_sizes; (void)n_in; (void)out_size; (void)d_ws; (void)ws_size;
    const float* q = (const float*)d_in[0];
    const float* k = (const float*)d_in[1];
    const float* v = (const float*)d_in[2];
    float* out  = (float*)d_out;                                   // [B,H,S,DK] first
    float* attn = out + (size_t)B_DIM * H_DIM * S_DIM * DK_DIM;    // then [B,H,S,S]
    dim3 grid(B_DIM * (S_DIM / TI));   // 256 workgroups: (batch, 32-row i-tile)
    sdpa_headsoftmax_fused<<<grid, 512, 0, stream>>>(q, k, v, out, attn);
}
